// GatedDeltaProduct_40097814675545
// MI455X (gfx1250) — matrix operand order repaired and checked by reading, not compiled, not run on hardware
//
#include <hip/hip_runtime.h>
#include <hip/hip_bf16.h>
#include <math.h>

#define HD    8      // num_heads
#define DH    128    // head_dim
#define NHH   2      // num_householder
#define HIDD  2048   // hidden_size
#define KDIM  1024   // KEY_DIM = HD*DH
#define CONVK 4
#define EPSF  1e-5f

typedef __attribute__((ext_vector_type(16))) _Float16 v16h;
typedef __attribute__((ext_vector_type(8)))  _Float16 v8h;
typedef __attribute__((ext_vector_type(8)))  float    v8f;
typedef int v4i_vs __attribute__((vector_size(16)));   // matches builtin param

union V16U { v16h v; v8h h[2]; };

// ---- async global->LDS (ASYNCcnt) path, guarded so either toolchain compiles
#if defined(__gfx1250__) && __has_builtin(__builtin_amdgcn_global_load_async_to_lds_b128)
#define USE_ASYNC_LDS 1
// Builtin signature (from clang diagnostic): (v4i AS1*, v4i AS3*, imm, imm).
// generic->AS1 is numerically identity; generic->AS3 is the architected
// low-32-bit truncation; go through
// integers to sidestep address-space cast legality.
#define ASYNC_CP(ldsp, gp)                                                      \
  __builtin_amdgcn_global_load_async_to_lds_b128(                               \
      (__attribute__((address_space(1))) v4i_vs*)(unsigned long long)(const void*)(gp), \
      (__attribute__((address_space(3))) v4i_vs*)(unsigned)(unsigned long long)(const void*)(ldsp), \
      0, 0)
#if __has_builtin(__builtin_amdgcn_s_wait_asynccnt)
#define WAIT_ASYNC(n) __builtin_amdgcn_s_wait_asynccnt(n)
#else
#define WAIT_ASYNC(n) asm volatile("s_wait_asynccnt %0" ::"i"(n) : "memory")
#endif
#else
#define USE_ASYNC_LDS 0
#endif

// ---------------------------------------------------------------- f32 -> f16
__global__ void k_f32_to_f16(const float* __restrict__ in,
                             _Float16* __restrict__ out, int n) {
  int i = blockIdx.x * blockDim.x + threadIdx.x;
  if (i < n) out[i] = (_Float16)in[i];
}

// -------------------------------------------------- WMMA GEMM: C = A * W^T
// A: [M][K] f16 row-major, W: [N][K] f16 row-major, C: [M][N] f32.
// Block = 256 thr = 8 waves, each wave one 16x16 tile; block covers 64x32.
__global__ __launch_bounds__(256) void k_wmma_gemm_nt(
    const _Float16* __restrict__ A, const _Float16* __restrict__ W,
    float* __restrict__ C, int M, int N, int K) {
  const int wave = threadIdx.x >> 5;
  const int lane = threadIdx.x & 31;
  const int g    = lane >> 4;       // half-wave group
  const int mn   = lane & 15;
  const int tileM = blockIdx.x * 64 + (wave & 3) * 16;
  const int tileN = blockIdx.y * 32 + (wave >> 2) * 16;
  const _Float16* arow = A + (size_t)(tileM + mn) * K;
  const _Float16* wrow = W + (size_t)(tileN + mn) * K;
  v8f acc = {};
  for (int k0 = 0; k0 < K; k0 += 32) {
    V16U a, b;
    // 16-bit A 16x32 layout: g=0 holds K[0..7]&[16..23], g=1 K[8..15]&[24..31]
    a.h[0] = *(const v8h*)(arow + k0 + 8 * g);
    a.h[1] = *(const v8h*)(arow + k0 + 16 + 8 * g);
    // B 32x16 layout: same as A: g=0 holds K[0..7]&[16..23], g=1 K[8..15]&[24..31]
    b.h[0] = *(const v8h*)(wrow + k0 + 8 * g);
    b.h[1] = *(const v8h*)(wrow + k0 + 16 + 8 * g);
    acc = __builtin_amdgcn_wmma_f32_16x16x32_f16(false, a.v, false, b.v,
                                                 (short)0, acc, false, false);
  }
  // C VGPR r: lanes 0-15 -> M=r, lanes 16-31 -> M=r+8; N = tileN + mn
  float* crow = C + (size_t)(tileM + 8 * g) * N + tileN + mn;
#pragma unroll
  for (int r = 0; r < 8; ++r) crow[(size_t)r * N] = acc[r];
}

// ------------------------------------------ beta / g (tiny projections, f32)
__global__ void k_small_proj(const float* __restrict__ x,
                             const float* __restrict__ Wb,
                             const float* __restrict__ Wa,
                             const float* __restrict__ Alog,
                             const float* __restrict__ dtb,
                             float* __restrict__ beta,
                             float* __restrict__ gdec) {
  const int m = blockIdx.x;
  const int t = threadIdx.x;
  const float* xr = x + (size_t)m * HIDD;
  if (t < 16) {
    const float* w = Wb + (size_t)t * HIDD;
    float s = 0.f;
    for (int k = 0; k < HIDD; ++k) s += xr[k] * w[k];
    beta[(size_t)m * 16 + t] = 2.f / (1.f + expf(-s));  // 2*sigmoid
  } else if (t < 24) {
    const int j = t - 16;
    const float* w = Wa + (size_t)j * HIDD;
    float s = 0.f;
    for (int k = 0; k < HIDD; ++k) s += xr[k] * w[k];
    float z = s + dtb[j];
    float sp = (z > 20.f) ? z : log1pf(expf(z));         // softplus
    gdec[(size_t)m * HD + j] = -expf(Alog[j]) * sp;
  }
}

// -------------------------------- depthwise causal conv1d (K=4) + SiLU
__global__ void k_dwconv_silu(const float* __restrict__ X,
                              const float* __restrict__ Wc,
                              float* __restrict__ Y, int B, int L, int C) {
  int idx = blockIdx.x * blockDim.x + threadIdx.x;
  int total = B * L * C;
  if (idx >= total) return;
  int c = idx % C;
  int l = (idx / C) % L;
  int b = idx / (C * L);
  const float* w = Wc + (size_t)c * CONVK;
  float acc = 0.f;
#pragma unroll
  for (int j = 0; j < CONVK; ++j) {
    int ls = l - (CONVK - 1) + j;
    if (ls >= 0) acc += X[((size_t)(b * L + ls)) * C + c] * w[j];
  }
  Y[idx] = acc / (1.f + expf(-acc));  // SiLU
}

// ------------------------------- L2-normalize each contiguous 128-float group
__global__ void k_l2norm128(float* __restrict__ D, int ngroups, float postscale) {
  int grp = blockIdx.x * (blockDim.x >> 5) + (threadIdx.x >> 5);
  if (grp >= ngroups) return;
  int lane = threadIdx.x & 31;
  float* p = D + (size_t)grp * DH;
  float v0 = p[lane], v1 = p[lane + 32], v2 = p[lane + 64], v3 = p[lane + 96];
  float s = v0 * v0 + v1 * v1 + v2 * v2 + v3 * v3;
  for (int o = 16; o > 0; o >>= 1) s += __shfl_xor(s, o, 32);
  float inv = postscale / fmaxf(sqrtf(s), 1e-12f);
  p[lane] = v0 * inv; p[lane + 32] = v1 * inv;
  p[lane + 64] = v2 * inv; p[lane + 96] = v3 * inv;
}

// ------------------------------------------- sequential gated delta scan
// One block per (b,h); 128x128 fp32 state resident in LDS for all T steps.
// k/v/q for step t+1 are prefetched into double-buffered LDS with
// GLOBAL_LOAD_ASYNC_TO_LDS_B128 (ASYNCcnt) while step t computes.
__global__ __launch_bounds__(256) void k_scan(
    const float* __restrict__ qn,    // (B,L,HD,DH)  q at last substep
    const float* __restrict__ kn,    // (B,T,HD,DH)
    const float* __restrict__ vn,    // (B,T,HD,DH)
    const float* __restrict__ gdec,  // (B,L,HD)     decay (substep 0 only)
    const float* __restrict__ beta,  // (B,T,HD)
    float* __restrict__ O,           // (B,L,HD,DH)
    float* __restrict__ stateOut,    // (B,HD,DH,DH)
    int Lc) {
  const int b = blockIdx.x >> 3;
  const int h = blockIdx.x & 7;
  const int T = Lc * NHH;
  __shared__ float st[DH][DH];
  __shared__ float kbuf[2][DH], vbuf[2][DH], qbuf[2][DH];
  __shared__ float vnew[DH];
  __shared__ float cpart[2][DH];
  const int tid  = threadIdx.x;
  const int v    = tid & (DH - 1);
  const int half = tid >> 7;
  float* sflat = &st[0][0];
  for (int i = tid; i < DH * DH; i += 256) sflat[i] = 0.f;

#if USE_ASYNC_LDS
  if (tid < 32) {  // wave 0: prefetch t = 0 (3 x 512B, one b128 per lane)
    const size_t b0 = ((size_t)(b * T) * HD + h) * DH;
    const size_t q0 = ((size_t)(b * Lc) * HD + h) * DH;
    ASYNC_CP(&kbuf[0][4 * tid], kn + b0 + 4 * tid);
    ASYNC_CP(&vbuf[0][4 * tid], vn + b0 + 4 * tid);
    ASYNC_CP(&qbuf[0][4 * tid], qn + q0 + 4 * tid);
  }
#endif
  __syncthreads();

  for (int t = 0; t < T; ++t) {
    const int l   = t >> 1;
    const int nh  = t & 1;
    const int buf = t & 1;

#if USE_ASYNC_LDS
    if (tid < 32) {  // wave 0: kick t+1 into other buffer, then wait for t
      if (t + 1 < T) {
        const int tn = t + 1;
        const size_t bn = ((size_t)(b * T + tn) * HD + h) * DH;
        const size_t qbn = ((size_t)(b * Lc + (tn >> 1)) * HD + h) * DH;
        ASYNC_CP(&kbuf[buf ^ 1][4 * tid], kn + bn + 4 * tid);
        ASYNC_CP(&vbuf[buf ^ 1][4 * tid], vn + bn + 4 * tid);
        ASYNC_CP(&qbuf[buf ^ 1][4 * tid], qn + qbn + 4 * tid);
        WAIT_ASYNC(3);  // in-order retirement: 3 oldest (= buffer `buf`) done
      } else {
        WAIT_ASYNC(0);
      }
    }
#else
    {
      const size_t base = ((size_t)(b * T + t) * HD + h) * DH;
      if (tid < DH) {
        kbuf[buf][tid] = kn[base + tid];
        vbuf[buf][tid] = vn[base + tid];
        qbuf[buf][tid] = qn[((size_t)(b * Lc + l) * HD + h) * DH + tid];
      }
    }
#endif
    if (nh == 0) {  // gate applied only at first householder substep
      const float gs = expf(gdec[(size_t)(b * Lc + l) * HD + h]);
      for (int i = tid; i < DH * DH; i += 256) sflat[i] *= gs;
    }
    __syncthreads();

    // corr[v] = sum_k k[k] * S[k][v]   (k-range split across the two halves)
    {
      float s = 0.f;
      const int k0 = half * 64;
#pragma unroll 8
      for (int kk = k0; kk < k0 + 64; ++kk) s += kbuf[buf][kk] * st[kk][v];
      cpart[half][v] = s;
    }
    __syncthreads();

    if (tid < DH) {
      const float bt = beta[(size_t)(b * T + t) * HD + h];
      vnew[tid] = (vbuf[buf][tid] - (cpart[0][tid] + cpart[1][tid])) * bt;
    }
    __syncthreads();

    // rank-1 update: S[k][v] += k[k]*vnew[v]
    for (int i = tid; i < DH * DH; i += 256) {
      const int kk = i >> 7, vv = i & (DH - 1);
      sflat[i] += kbuf[buf][kk] * vnew[vv];
    }
    __syncthreads();

    if (nh == 1) {  // output only at last householder substep
      float s2 = 0.f;
      const int k0 = half * 64;
#pragma unroll 8
      for (int kk = k0; kk < k0 + 64; ++kk) s2 += qbuf[buf][kk] * st[kk][v];
      cpart[half][v] = s2;
      __syncthreads();
      if (tid < DH)
        O[((size_t)(b * Lc + l) * HD + h) * DH + tid] =
            cpart[0][tid] + cpart[1][tid];
      __syncthreads();
    }
  }
  float* so = stateOut + (size_t)(b * HD + h) * DH * DH;
  for (int i = tid; i < DH * DH; i += 256) so[i] = sflat[i];
}

// ----------------------- gated RMSNorm: o*rsqrt(mean o^2+eps)*w*silu(gg) -> f16
__global__ void k_gated_rmsnorm(const float* __restrict__ O,
                                const float* __restrict__ Gg,
                                const float* __restrict__ w,
                                _Float16* __restrict__ Oh, int ngroups) {
  int grp = blockIdx.x * (blockDim.x >> 5) + (threadIdx.x >> 5);
  if (grp >= ngroups) return;
  int lane = threadIdx.x & 31;
  const float* p  = O + (size_t)grp * DH;
  const float* gp = Gg + (size_t)grp * DH;
  float vv[4]; float s = 0.f;
#pragma unroll
  for (int i = 0; i < 4; ++i) { vv[i] = p[lane + 32 * i]; s += vv[i] * vv[i]; }
  for (int o = 16; o > 0; o >>= 1) s += __shfl_xor(s, o, 32);
  float r = rsqrtf(s * (1.f / DH) + EPSF);
#pragma unroll
  for (int i = 0; i < 4; ++i) {
    int d = lane + 32 * i;
    float gg = gp[d];
    float sg = gg / (1.f + expf(-gg));  // silu
    Oh[(size_t)grp * DH + d] = (_Float16)(vv[i] * r * w[d] * sg);
  }
}

// ======================================================================
extern "C" void kernel_launch(void* const* d_in, const int* in_sizes, int n_in,
                              void* d_out, int out_size, void* d_ws, size_t ws_size,
                              hipStream_t stream) {
  const float* x      = (const float*)d_in[0];
  const float* Wq     = (const float*)d_in[1];
  const float* Wk     = (const float*)d_in[2];
  const float* Wv     = (const float*)d_in[3];
  const float* Wb     = (const float*)d_in[4];
  const float* Wa     = (const float*)d_in[5];
  const float* Wg     = (const float*)d_in[6];
  const float* Wo     = (const float*)d_in[7];
  const float* A_log  = (const float*)d_in[8];
  const float* dt_b   = (const float*)d_in[9];
  const float* conv_q = (const float*)d_in[10];
  const float* conv_k = (const float*)d_in[11];
  const float* conv_v = (const float*)d_in[12];
  const float* onw    = (const float*)d_in[13];

  const int L  = 1024;
  const int BL = in_sizes[0] / HIDD;   // B*L
  const int B  = BL / L;
  const int BT = BL * NHH;

  // ---- workspace bump allocator
  char* wp = (char*)d_ws;
  auto alloc = [&](size_t bytes) -> void* {
    void* p = (void*)wp;
    wp += (bytes + 255) & ~(size_t)255;
    return p;
  };
  _Float16* Xh   = (_Float16*)alloc((size_t)BL * HIDD * 2);
  _Float16* Wqh  = (_Float16*)alloc((size_t)KDIM * HIDD * 2);
  _Float16* Wkh  = (_Float16*)alloc((size_t)KDIM * NHH * HIDD * 2);
  _Float16* Wvh  = (_Float16*)alloc((size_t)KDIM * NHH * HIDD * 2);
  _Float16* Wgh  = (_Float16*)alloc((size_t)KDIM * HIDD * 2);
  _Float16* Woh  = (_Float16*)alloc((size_t)HIDD * KDIM * 2);
  float* Qpre    = (float*)alloc((size_t)BL * KDIM * 4);
  float* Kpre    = (float*)alloc((size_t)BL * KDIM * NHH * 4);
  float* Vpre    = (float*)alloc((size_t)BL * KDIM * NHH * 4);
  float* Gg      = (float*)alloc((size_t)BL * KDIM * 4);
  float* Qc      = (float*)alloc((size_t)BL * KDIM * 4);
  float* Kc      = (float*)alloc((size_t)BL * KDIM * NHH * 4);
  float* Vc      = (float*)alloc((size_t)BL * KDIM * NHH * 4);
  float* betab   = (float*)alloc((size_t)BL * 16 * 4);
  float* gdec    = (float*)alloc((size_t)BL * HD * 4);
  float* Obuf    = (float*)alloc((size_t)BL * KDIM * 4);
  _Float16* Oh   = (_Float16*)alloc((size_t)BL * KDIM * 2);

  float* outY     = (float*)d_out;                       // (B,L,HID)
  float* outState = (float*)d_out + (size_t)BL * HIDD;   // (B,HD,DH,DH)

  auto cvt = [&](const float* src, _Float16* dst, size_t n) {
    k_f32_to_f16<<<(n + 255) / 256, 256, 0, stream>>>(src, dst, (int)n);
  };

  // ---- 1) f32 -> f16 copies for WMMA operands
  cvt(x,  Xh,  (size_t)BL * HIDD);
  cvt(Wq, Wqh, (size_t)KDIM * HIDD);
  cvt(Wk, Wkh, (size_t)KDIM * NHH * HIDD);
  cvt(Wv, Wvh, (size_t)KDIM * NHH * HIDD);
  cvt(Wg, Wgh, (size_t)KDIM * HIDD);
  cvt(Wo, Woh, (size_t)HIDD * KDIM);

  // ---- 2) projection GEMMs (WMMA f32 <- f16xf16)
  {
    dim3 blk(256);
    dim3 gq(BL / 64, KDIM / 32);
    dim3 gkv(BL / 64, (KDIM * NHH) / 32);
    k_wmma_gemm_nt<<<gq,  blk, 0, stream>>>(Xh, Wqh, Qpre, BL, KDIM, HIDD);
    k_wmma_gemm_nt<<<gkv, blk, 0, stream>>>(Xh, Wkh, Kpre, BL, KDIM * NHH, HIDD);
    k_wmma_gemm_nt<<<gkv, blk, 0, stream>>>(Xh, Wvh, Vpre, BL, KDIM * NHH, HIDD);
    k_wmma_gemm_nt<<<gq,  blk, 0, stream>>>(Xh, Wgh, Gg,   BL, KDIM, HIDD);
  }

  // ---- 3) beta / g (fp32 path: feeds exp/sigmoid)
  k_small_proj<<<BL, 32, 0, stream>>>(x, Wb, Wa, A_log, dt_b, betab, gdec);

  // ---- 4) depthwise causal conv + SiLU
  k_dwconv_silu<<<((size_t)BL * KDIM + 255) / 256, 256, 0, stream>>>(
      Qpre, conv_q, Qc, B, L, KDIM);
  k_dwconv_silu<<<((size_t)BL * KDIM * NHH + 255) / 256, 256, 0, stream>>>(
      Kpre, conv_k, Kc, B, L, KDIM * NHH);
  k_dwconv_silu<<<((size_t)BL * KDIM * NHH + 255) / 256, 256, 0, stream>>>(
      Vpre, conv_v, Vc, B, L, KDIM * NHH);

  // ---- 5) L2 norms: q/=sqrt(DH) after norm; k normed only
  {
    int gq = BL * HD;         // q groups
    int gk = BT * HD;         // k groups
    k_l2norm128<<<(gq + 7) / 8, 256, 0, stream>>>(Qc, gq, rsqrtf((float)DH));
    k_l2norm128<<<(gk + 7) / 8, 256, 0, stream>>>(Kc, gk, 1.0f);
  }

  // ---- 6) sequential scan, one block per (b,h); final state -> d_out tail
  k_scan<<<B * HD, 256, 0, stream>>>(Qc, Kc, Vc, gdec, betab, Obuf, outState, L);

  // ---- 7) gated RMSNorm -> f16
  k_gated_rmsnorm<<<(BL * HD + 7) / 8, 256, 0, stream>>>(Obuf, Gg, onw, Oh, BL * HD);

  // ---- 8) output GEMM: (B,L,KDIM) @ Wo^T -> (B,L,HID) fp32 into d_out
  {
    dim3 blk(256);
    dim3 go(BL / 64, HIDD / 32);
    k_wmma_gemm_nt<<<go, blk, 0, stream>>>(Oh, Woh, outY, BL, HIDD, KDIM);
  }
}
